// MambaEncoder_66168266162295
// MI455X (gfx1250) — matrix operand order repaired and checked by reading, not compiled, not run on hardware
//
#include <hip/hip_runtime.h>
#include <hip/hip_bf16.h>
#include <math.h>

// ---------------- problem constants ----------------
#define BB 32
#define TT 256
#define FF 128
#define RR 32
#define DM 1024
#define DSTATE 16
#define DCONV 4
#define DI 2048
#define DTR 64
#define NL 3
#define TRH (48.0f / 31.0f)   // tau_ref/3600, constant

typedef __attribute__((ext_vector_type(16))) _Float16 v16h;
typedef __attribute__((ext_vector_type(8)))  _Float16 v8h;
typedef __attribute__((ext_vector_type(8)))  float    v8f;
typedef __attribute__((ext_vector_type(4)))  unsigned int u32x4;
typedef __attribute__((ext_vector_type(8)))  int      i32x8;
typedef __attribute__((ext_vector_type(4)))  int      i32x4;

#if defined(__has_builtin)
#if __has_builtin(__builtin_amdgcn_tensor_load_to_lds) && __has_builtin(__builtin_amdgcn_s_wait_tensorcnt)
#define HAVE_TDM 1
#endif
#endif

__device__ __forceinline__ float sp_f(float x) {            // softplus
    return (x > 20.0f) ? x : log1pf(expf(x));
}
__device__ __forceinline__ float silu_f(float x) {
    return x / (1.0f + expf(-x));
}

#if defined(HAVE_TDM)
// 2-D f32 tile load via Tensor Data Mover: rows x width from g (row stride
// `stride` elems) into LDS at byte offset lds_b (rows stored compacted).
__device__ __forceinline__ void tdm_load_2d_f32(unsigned lds_b, const float* g,
                                                int tile_w, int tile_h,
                                                int stride, int tdim0, int tdim1) {
    const unsigned long long ga = (unsigned long long)(uintptr_t)g;
    u32x4 g0 = { 1u,                                        // count=1, user mode
                 lds_b,                                     // lds_addr
                 (unsigned)(ga & 0xffffffffull),            // global_addr[31:0]
                 (unsigned)((ga >> 32) & 0x01ffffffull) | (2u << 30) }; // hi | type=2
    i32x8 g1 = { (int)(2u << 16),                           // data_size=4B
                 (int)(((unsigned)tdim0 & 0xffffu) << 16),  // tensor_dim0 lo
                 (int)((((unsigned)tdim0 >> 16) & 0xffffu) | (((unsigned)tdim1 & 0xffffu) << 16)),
                 (int)(((((unsigned)tdim1) >> 16) & 0xffffu) | (((unsigned)tile_w & 0xffffu) << 16)),
                 (int)((unsigned)tile_h & 0xffffu),         // tile_dim1 (tile_dim2=0)
                 (int)stride,                               // tensor_dim0_stride lo
                 0, 0 };
    i32x4 g2 = { 1, 0, 0, 0 };                              // tensor_dim2=1
    i32x4 g3 = { 0, 0, 0, 0 };
#if defined(__clang_major__) && (__clang_major__ >= 23)
    i32x8 g4 = { 0, 0, 0, 0, 0, 0, 0, 0 };
    __builtin_amdgcn_tensor_load_to_lds(g0, g1, g2, g3, g4, 0);
#else
    __builtin_amdgcn_tensor_load_to_lds(g0, g1, g2, g3, 0);
#endif
}
#endif

// ---------------- 1) per-feature masked mean ----------------
__global__ __launch_bounds__(256) void k_mean(const float* __restrict__ x,
                                              const float* __restrict__ mask,
                                              float* __restrict__ x_mean) {
    const int f = blockIdx.x;             // 0..127
    const int tid = threadIdx.x;
    __shared__ float s1[256], s2[256];
    float a = 0.f, m = 0.f;
    for (int bt = tid; bt < BB * TT; bt += 256) {
        float mv = mask[(size_t)bt * FF + f];
        a += x[(size_t)bt * FF + f] * mv;
        m += mv;
    }
    s1[tid] = a; s2[tid] = m; __syncthreads();
    for (int s = 128; s > 0; s >>= 1) {
        if (tid < s) { s1[tid] += s1[tid + s]; s2[tid] += s2[tid + s]; }
        __syncthreads();
    }
    if (tid == 0) x_mean[f] = s1[0] / fmaxf(s2[0], 1.0f);
}

// ---------------- 2) cumsum of tau -> t_hours ----------------
__global__ __launch_bounds__(256) void k_cumsum(const float* __restrict__ tau,
                                                float* __restrict__ t_hours) {
    const int b = blockIdx.x, tid = threadIdx.x;
    __shared__ float sh[256];
    float v = tau[b * TT + tid];
    sh[tid] = v; __syncthreads();
    for (int off = 1; off < 256; off <<= 1) {
        float add = (tid >= off) ? sh[tid - off] : 0.f;
        __syncthreads();
        v += add; sh[tid] = v;
        __syncthreads();
    }
    t_hours[b * TT + tid] = v / 3600.0f;
}

// ---------------- 3) imputation -> x_hat ----------------
__global__ __launch_bounds__(256) void k_xhat(const float* __restrict__ x,
                                              const float* __restrict__ tau,
                                              const float* __restrict__ mask,
                                              const float* __restrict__ log_w,
                                              const float* __restrict__ x_mean,
                                              float* __restrict__ x_hat) {
    const size_t idx = (size_t)blockIdx.x * 256 + threadIdx.x;
    if (idx >= (size_t)BB * TT * FF) return;
    const int f = (int)(idx % FF);
    const size_t bt = idx / FF;
    const float th = fmaxf(tau[bt] / 3600.0f, 1e-4f);
    const float g = expf(-sp_f(log_w[f]) * th);
    const float xv = x[idx], mv = mask[idx];
    x_hat[idx] = mv * xv + (1.0f - mv) * (g * xv + (1.0f - g) * x_mean[f]);
}

// ---------------- 4) kernel-weighted aggregation ----------------
__global__ __launch_bounds__(128) void k_agg(const float* __restrict__ x_hat,
                                             const float* __restrict__ mask,
                                             const float* __restrict__ t_hours,
                                             const float* __restrict__ log_kernel,
                                             float* __restrict__ lam,
                                             float* __restrict__ sigma,
                                             float* __restrict__ gamma) {
    const int br = blockIdx.x;            // b*R + r
    const int b = br >> 5, r = br & 31;
    const int f = threadIdx.x;            // 0..127
    __shared__ float th[TT];
    for (int i = f; i < TT; i += 128) th[i] = t_hours[b * TT + i];
    __syncthreads();
    const float ref = 48.0f * (float)r / 31.0f;
    const float kap = sp_f(log_kernel[f]);
    float lamA = 0.f, sN = 0.f, hS = 0.f, gN = 0.f;
    for (int t = 0; t < TT; ++t) {
        const float d = th[t] - ref;
        const float e = d * d * kap;
        const size_t gi = ((size_t)(b * TT + t)) * FF + f;
        const float m = mask[gi];
        const float xv = x_hat[gi];
        const float wl = expf(-e) * m;
        const float wh = expf(-10.0f * e) * m;
        lamA += wl; sN += wl * xv; hS += wh; gN += wh * xv;
    }
    const size_t o = (size_t)br * FF + f;
    lam[o]   = lamA;
    sigma[o] = sN / fmaxf(lamA, 1.0f);
    gamma[o] = gN / fmaxf(hS, 1.0f);
}

// ---------------- 5) rho mixing + feature assembly ----------------
__global__ __launch_bounds__(128) void k_feat(const float* __restrict__ lam,
                                              const float* __restrict__ sigma,
                                              const float* __restrict__ gamma,
                                              const float* __restrict__ rho,
                                              float* __restrict__ feat) {
    const int row = blockIdx.x;           // 0..1023 (b*R+r)
    const int j = threadIdx.x;            // 0..127
    __shared__ float lamS[FF], lsS[FF];
    const size_t o = (size_t)row * FF + j;
    lamS[j] = lam[o];
    lsS[j]  = lamS[j] * sigma[o];
    __syncthreads();
    float num = 0.f, den = 0.f;
    for (int k = 0; k < FF; ++k) {
        const float rk = rho[k * FF + j];
        num += rk * lsS[k];
        den += rk * lamS[k];
    }
    den = fmaxf(den, 1.0f);
    const float chi = num / den;
    feat[(size_t)row * (3 * FF) + j]          = lamS[j];
    feat[(size_t)row * (3 * FF) + FF + j]     = chi;
    feat[(size_t)row * (3 * FF) + 2 * FF + j] = gamma[o] - chi;
}

// ---------------- 6) generic WMMA f16 GEMM: C = act(A*B + bias) ----------------
// Block tile 64x128, 4 waves in a 2x2 grid, each wave 32x64 = 2x4 wmma tiles.
// Both A (64x32) and B (32x128) f32 tiles are DMA'd into LDS by the Tensor
// Data Mover (double-duty prefetch: slab i+1 is issued before the WMMA work
// of slab i and completed with s_wait_tensorcnt after it). A convert pass
// produces f16 fragment-major layouts so each lane fragment is 2x ds_load_b128.
// act: 0 none, 1 softplus, 2 tanh
__global__ __launch_bounds__(128) void k_gemm_wmma(const float* __restrict__ A, int lda,
                                                   const float* __restrict__ Bw, int ldb,
                                                   const float* __restrict__ bias,
                                                   float* __restrict__ C, int ldc,
                                                   int M, int N, int K, int act) {
    __shared__ __align__(16) float    AsF[64][32];     // raw f32 A tile (TDM dest)
    __shared__ __align__(16) float    BsF[32][128];    // raw f32 B tile (TDM dest)
    __shared__ __align__(16) _Float16 As_p[64][40];    // [row][k], k contiguous
    __shared__ __align__(16) _Float16 Bs_t[128][40];   // [col][k], k contiguous
    const int tid  = threadIdx.x;
    const int lane = tid & 31;
    const int wave = tid >> 5;            // 0..3
    const int wr = wave >> 1, wc = wave & 1;
    const int half = lane >> 4;           // lane group
    const int l15  = lane & 15;
    const int rowBase = blockIdx.y * 64;
    const int colBase = blockIdx.x * 128;

    v8f acc[2][4];
#pragma unroll
    for (int i = 0; i < 2; ++i)
#pragma unroll
        for (int j = 0; j < 4; ++j)
#pragma unroll
            for (int e = 0; e < 8; ++e) acc[i][j][e] = 0.0f;

    // stage(kk): bring A(64x32)@kk and B(32x128)@kk f32 tiles into LDS
    auto stage = [&](int kk) {
#if defined(HAVE_TDM)
        if (wave == 0) {
            tdm_load_2d_f32((unsigned)(uintptr_t)(&AsF[0][0]),
                            A + (size_t)rowBase * lda + kk,
                            /*tile_w=*/32, /*tile_h=*/64, /*stride=*/lda,
                            /*tdim0=*/K - kk, /*tdim1=*/M - rowBase);
            tdm_load_2d_f32((unsigned)(uintptr_t)(&BsF[0][0]),
                            Bw + (size_t)kk * ldb + colBase,
                            /*tile_w=*/128, /*tile_h=*/32, /*stride=*/ldb,
                            /*tdim0=*/N - colBase, /*tdim1=*/K - kk);
        }
#else
#pragma unroll
        for (int it = 0; it < 16; ++it) {            // A: 64*32 / 128 threads
            const int e = tid + it * 128;
            const int r = e >> 5, c = e & 31;
            const int gr = rowBase + r, gc = kk + c;
            const int grc = gr < M - 1 ? gr : M - 1;
            const int gcc = gc < K - 1 ? gc : K - 1;
            float v = A[(size_t)grc * lda + gcc];
            AsF[r][c] = (gr < M && gc < K) ? v : 0.0f;
        }
#pragma unroll
        for (int it = 0; it < 32; ++it) {            // B: 32*128 / 128 threads
            const int e = tid + it * 128;
            const int r = e >> 7, c = e & 127;
            const int gr = kk + r, gc = colBase + c;
            const int grc = gr < K - 1 ? gr : K - 1;
            const int gcc = gc < N - 1 ? gc : N - 1;
            float v = Bw[(size_t)grc * ldb + gcc];
            BsF[r][c] = (gr < K && gc < N) ? v : 0.0f;
        }
#endif
    };
    auto stage_sync = [&]() {       // publish staged tiles to all waves
#if defined(HAVE_TDM)
        if (wave == 0) __builtin_amdgcn_s_wait_tensorcnt(0);
#endif
        __syncthreads();
    };

    stage(0);
    stage_sync();

    for (int k0 = 0; k0 < K; k0 += 32) {
        // ---- convert raw f32 tiles into f16 fragment-major layouts ----
#pragma unroll
        for (int it = 0; it < 16; ++it) {            // A convert, k contiguous
            const int e = tid + it * 128;
            const int r = e >> 5, c = e & 31;
            float v = AsF[r][c];
            if (rowBase + r >= M || k0 + c >= K) v = 0.0f;
            As_p[r][c] = (_Float16)v;
        }
#pragma unroll
        for (int it = 0; it < 32; ++it) {            // B transpose-convert
            const int e = tid + it * 128;
            const int r = e >> 7, c = e & 127;
            float v = BsF[r][c];
            if (colBase + c >= N || k0 + r >= K) v = 0.0f;
            Bs_t[c][r] = (_Float16)v;
        }
        __syncthreads();

        // ---- async prefetch of next K-slab overlaps the WMMA work ----
        if (k0 + 32 < K) stage(k0 + 32);

        // ---- WMMA: 2x4 tiles per wave, contiguous b128 fragment loads ----
        v16h afr[2], bfr[4];
#pragma unroll
        for (int ti = 0; ti < 2; ++ti) {
            const v8h* pa = (const v8h*)(&As_p[wr * 32 + ti * 16 + l15][half * 16]);
            afr[ti] = __builtin_shufflevector(pa[0], pa[1],
                0, 1, 2, 3, 4, 5, 6, 7, 8, 9, 10, 11, 12, 13, 14, 15);
        }
#pragma unroll
        for (int tj = 0; tj < 4; ++tj) {
            const v8h* pb = (const v8h*)(&Bs_t[wc * 64 + tj * 16 + l15][half * 16]);
            bfr[tj] = __builtin_shufflevector(pb[0], pb[1],
                0, 1, 2, 3, 4, 5, 6, 7, 8, 9, 10, 11, 12, 13, 14, 15);
        }
#pragma unroll
        for (int ti = 0; ti < 2; ++ti)
#pragma unroll
            for (int tj = 0; tj < 4; ++tj)
                acc[ti][tj] = __builtin_amdgcn_wmma_f32_16x16x32_f16(
                    false, afr[ti], false, bfr[tj], (short)0, acc[ti][tj], false, false);

        stage_sync();
    }

    // ---- epilogue: bias + activation, guarded store ----
#pragma unroll
    for (int ti = 0; ti < 2; ++ti)
#pragma unroll
        for (int tj = 0; tj < 4; ++tj) {
            const int m0 = rowBase + wr * 32 + ti * 16;
            const int col = colBase + wc * 64 + tj * 16 + l15;
#pragma unroll
            for (int e = 0; e < 8; ++e) {
                const int row = m0 + half * 8 + e;
                if (row < M && col < N) {
                    float v = acc[ti][tj][e];
                    if (bias) v += bias[col];
                    if (act == 1) v = sp_f(v);
                    else if (act == 2) v = tanhf(v);
                    C[(size_t)row * ldc + col] = v;
                }
            }
        }
}

// ---------------- 7) layer norm (block per row), optional nan_to_num ----------------
__global__ __launch_bounds__(256) void k_ln(const float* __restrict__ in,
                                            float* __restrict__ out,
                                            const float* __restrict__ g,
                                            const float* __restrict__ b,
                                            int C, int nan_fix) {
    const int row = blockIdx.x, tid = threadIdx.x;
    __shared__ float s1[256], s2[256];
    float a = 0.f, q = 0.f;
    for (int c = tid; c < C; c += 256) {
        const float v = in[(size_t)row * C + c];
        a += v; q += v * v;
    }
    s1[tid] = a; s2[tid] = q; __syncthreads();
    for (int s = 128; s > 0; s >>= 1) {
        if (tid < s) { s1[tid] += s1[tid + s]; s2[tid] += s2[tid + s]; }
        __syncthreads();
    }
    const float mean = s1[0] / (float)C;
    float var = s2[0] / (float)C - mean * mean;
    var = fmaxf(var, 0.0f);
    const float inv = rsqrtf(var + 1e-5f);
    for (int c = tid; c < C; c += 256) {
        float v = (in[(size_t)row * C + c] - mean) * inv * g[c] + b[c];
        if (nan_fix) {
            if (v != v) v = 0.0f;
            v = fminf(fmaxf(v, -3.4028235e38f), 3.4028235e38f);
        }
        out[(size_t)row * C + c] = v;
    }
}

// ---------------- 8) constant time-embedding vector te[1024] ----------------
__global__ __launch_bounds__(1024) void k_te(const float* __restrict__ W1,
                                             const float* __restrict__ b1,
                                             const float* __restrict__ W2,
                                             const float* __restrict__ b2,
                                             float* __restrict__ te) {
    const int j = threadIdx.x;
    __shared__ float s[DM];
    const float c = log1pf(TRH);
    s[j] = silu_f(c * W1[j] + b1[j]);
    __syncthreads();
    float acc = b2[j];
    for (int k = 0; k < DM; ++k) acc += s[k] * W2[k * DM + j];
    te[j] = acc;
}

// ---------------- 9) build [h | te] concat ----------------
__global__ __launch_bounds__(256) void k_cat2(const float* __restrict__ h,
                                              const float* __restrict__ te,
                                              float* __restrict__ cat2) {
    const size_t idx = (size_t)blockIdx.x * 256 + threadIdx.x;
    if (idx >= (size_t)BB * RR * 2 * DM) return;
    const int c = (int)(idx % (2 * DM));
    const size_t row = idx / (2 * DM);
    cat2[idx] = (c < DM) ? h[row * DM + c] : te[c - DM];
}

// ---------------- 10) depthwise causal conv + silu ----------------
__global__ __launch_bounds__(256) void k_conv(const float* __restrict__ xz,   // (rows,4096), xs = [:, :2048]
                                              const float* __restrict__ conv_w,
                                              const float* __restrict__ conv_b,
                                              float* __restrict__ xc) {
    const size_t idx = (size_t)blockIdx.x * 256 + threadIdx.x;
    if (idx >= (size_t)BB * RR * DI) return;
    const int d = (int)(idx % DI);
    const int row = (int)(idx / DI);
    const int b = row >> 5, r = row & 31;
    float acc = conv_b[d];
#pragma unroll
    for (int k = 0; k < DCONV; ++k) {
        const int rr = r - (DCONV - 1) + k;
        if (rr >= 0)
            acc += xz[(size_t)(b * RR + rr) * (2 * DI) + d] * conv_w[d * DCONV + k];
    }
    xc[(size_t)row * DI + d] = silu_f(acc);
}

// ---------------- 11) selective-state scan ----------------
__global__ __launch_bounds__(256) void k_scan(const float* __restrict__ dbc,   // (rows,96)
                                              const float* __restrict__ dtb,   // (rows,2048) softplus'd
                                              const float* __restrict__ xc,    // (rows,2048)
                                              const float* __restrict__ xz,    // z = [:, 2048:4096]
                                              const float* __restrict__ A_log, // (2048,16)
                                              const float* __restrict__ Dp,    // (2048)
                                              float* __restrict__ ybuf) {      // (rows,2048)
    const int b = blockIdx.y;
    const int d = blockIdx.x * 256 + threadIdx.x;   // 0..2047
    __shared__ float bc[2 * DSTATE];
    float aNeg[DSTATE], st[DSTATE];
#pragma unroll
    for (int s = 0; s < DSTATE; ++s) {
        aNeg[s] = -expf(A_log[(size_t)d * DSTATE + s]);
        st[s] = 0.0f;
    }
    const float Dv = Dp[d];
    for (int r = 0; r < RR; ++r) {
        const size_t row = (size_t)(b * RR + r);
        if (threadIdx.x < 2 * DSTATE)
            bc[threadIdx.x] = dbc[row * 96 + DTR + threadIdx.x];
        __syncthreads();
        const float dtv = dtb[row * DI + d];
        const float xcv = xc[row * DI + d];
        const float zv  = xz[row * (2 * DI) + DI + d];
        float y = 0.0f;
#pragma unroll
        for (int s = 0; s < DSTATE; ++s) {
            const float dA = expf(dtv * aNeg[s]);
            st[s] = dA * st[s] + dtv * bc[s] * xcv;
            y += st[s] * bc[DSTATE + s];
        }
        y = (y + Dv * xcv) * silu_f(zv);
        ybuf[row * DI + d] = y;
        __syncthreads();
    }
}

// ---------------- 12) gated residual: h += hm * exp(-softplus(log_decay)*trh) ----------------
__global__ __launch_bounds__(256) void k_resid(float* __restrict__ h,
                                               const float* __restrict__ hm,
                                               const float* __restrict__ log_decay) {
    const size_t idx = (size_t)blockIdx.x * 256 + threadIdx.x;
    if (idx >= (size_t)BB * RR * DM) return;
    const int d = (int)(idx % DM);
    h[idx] += hm[idx] * expf(-sp_f(log_decay[d]) * TRH);
}

// ---------------- 13) attention scores: scores[row] = dot(tanh_tmp[row,:], pool_u) ----------------
__global__ __launch_bounds__(128) void k_scores(const float* __restrict__ tmp, // (rows,512)
                                                const float* __restrict__ u,   // (512)
                                                float* __restrict__ scores) {
    const int row = blockIdx.x, tid = threadIdx.x;
    __shared__ float sh[128];
    float a = 0.f;
    for (int j = tid; j < DM / 2; j += 128) a += tmp[(size_t)row * (DM / 2) + j] * u[j];
    sh[tid] = a; __syncthreads();
    for (int s = 64; s > 0; s >>= 1) {
        if (tid < s) sh[tid] += sh[tid + s];
        __syncthreads();
    }
    if (tid == 0) scores[row] = sh[0];
}

// ---------------- 14) pooling -> cat4 (B, 4*DM) ----------------
__global__ __launch_bounds__(256) void k_pool(const float* __restrict__ hf,     // (rows,1024)
                                              const float* __restrict__ scores, // (rows)
                                              float* __restrict__ cat4) {       // (B,4096)
    const int b = blockIdx.x, tid = threadIdx.x;
    __shared__ float sc[RR];
    if (tid < RR) sc[tid] = scores[b * RR + tid];
    __syncthreads();
    float mx = -3.4e38f;
    for (int r = 0; r < RR; ++r) mx = fmaxf(mx, sc[r]);
    float se = 0.f;
    for (int r = 0; r < RR; ++r) se += expf(sc[r] - mx);
    for (int c = tid; c < DM; c += 256) {
        const float last = hf[(size_t)(b * RR + RR - 1) * DM + c];
        float mean = 0.f, mxv = -3.4e38f, attn = 0.f;
        for (int r = 0; r < RR; ++r) {
            const float v = hf[(size_t)(b * RR + r) * DM + c];
            mean += v; mxv = fmaxf(mxv, v);
            attn += (expf(sc[r] - mx) / se) * v;
        }
        cat4[(size_t)b * (4 * DM) + c]           = last;
        cat4[(size_t)b * (4 * DM) + DM + c]      = mean / (float)RR;
        cat4[(size_t)b * (4 * DM) + 2 * DM + c]  = mxv;
        cat4[(size_t)b * (4 * DM) + 3 * DM + c]  = attn;
    }
}

// ================================================================
extern "C" void kernel_launch(void* const* d_in, const int* in_sizes, int n_in,
                              void* d_out, int out_size, void* d_ws, size_t ws_size,
                              hipStream_t stream) {
    (void)in_sizes; (void)n_in; (void)out_size; (void)ws_size;
    // -------- inputs (setup_inputs dict order, params flattened) --------
    const float* x        = (const float*)d_in[0];
    const float* tau      = (const float*)d_in[1];
    const float* mask     = (const float*)d_in[2];
    const float* log_w    = (const float*)d_in[3];
    const float* log_kern = (const float*)d_in[4];
    const float* rho      = (const float*)d_in[5];
    const float* in_W     = (const float*)d_in[6];
    const float* in_b     = (const float*)d_in[7];
    const float* in_ln_g  = (const float*)d_in[8];
    const float* in_ln_b  = (const float*)d_in[9];
    const float* gap_W1   = (const float*)d_in[10];
    const float* gap_b1   = (const float*)d_in[11];
    const float* gap_W2   = (const float*)d_in[12];
    const float* gap_b2   = (const float*)d_in[13];
    const float* fuse_W   = (const float*)d_in[14];
    const float* fuse_b   = (const float*)d_in[15];
    // layers: 16 + 12*l + {0..11}
    const int LBASE = 16, LSTRIDE = 12;
    const float* norm_g      = (const float*)d_in[LBASE + NL * LSTRIDE + 0];
    const float* norm_b      = (const float*)d_in[LBASE + NL * LSTRIDE + 1];
    const float* pool_W      = (const float*)d_in[LBASE + NL * LSTRIDE + 2];
    const float* pool_Wb     = (const float*)d_in[LBASE + NL * LSTRIDE + 3];
    const float* pool_u      = (const float*)d_in[LBASE + NL * LSTRIDE + 4];
    const float* pool_proj   = (const float*)d_in[LBASE + NL * LSTRIDE + 5];
    const float* pool_proj_b = (const float*)d_in[LBASE + NL * LSTRIDE + 6];
    float* out = (float*)d_out;

    // -------- workspace layout (floats), with deliberate aliasing --------
    float* W = (float*)d_ws;
    float* t_hours = W;                    // 8192
    float* x_mean  = W + 8192;             // 128
    float* x_hat   = W + 8320;             // 1048576
    float* lam     = W + 1056896;          // 131072
    float* sigma   = W + 1187968;          // 131072
    float* gamma   = W + 1319040;          // 131072
    float* feat    = W + 1450112;          // 393216
    float* h       = W + 1843328;          // 1048576
    float* hn      = W + 2891904;          // 1048576 (also hf)
    float* te      = W + 3940480;          // 1024
    float* xz      = W + 3941504;          // 4194304 (also cat2 / tanh-tmp)
    float* xc      = W + 8135808;          // 2097152 (also cat4)
    float* dbc     = W + 10232960;         // 98304   (also scores)
    float* dtb     = W + 10331264;         // 2097152
    float* ybuf    = W + 12428416;         // 2097152
    float* hm      = W + 14525568;         // 1048576 (also hpre)

    const int ROWS = BB * RR;              // 1024
    auto cdiv = [](int a, int b) { return (a + b - 1) / b; };
    auto gemm = [&](const float* A, int lda, const float* Bw, int ldb,
                    const float* bias, float* C, int ldc, int M, int N, int K, int act) {
        dim3 g(cdiv(N, 128), cdiv(M, 64));
        k_gemm_wmma<<<g, 128, 0, stream>>>(A, lda, Bw, ldb, bias, C, ldc, M, N, K, act);
    };

    // ---- stage 1: imputation + aggregation ----
    k_mean<<<FF, 256, 0, stream>>>(x, mask, x_mean);
    k_cumsum<<<BB, 256, 0, stream>>>(tau, t_hours);
    k_xhat<<<cdiv(BB * TT * FF, 256), 256, 0, stream>>>(x, tau, mask, log_w, x_mean, x_hat);
    k_agg<<<BB * RR, 128, 0, stream>>>(x_hat, mask, t_hours, log_kern, lam, sigma, gamma);
    k_feat<<<ROWS, 128, 0, stream>>>(lam, sigma, gamma, rho, feat);

    // ---- stage 2: input projection + time-embedding fuse ----
    gemm(feat, 3 * FF, in_W, DM, in_b, hm, DM, ROWS, DM, 3 * FF, 0);      // hpre (alias hm)
    k_ln<<<ROWS, 256, 0, stream>>>(hm, h, in_ln_g, in_ln_b, DM, 0);
    k_te<<<1, DM, 0, stream>>>(gap_W1, gap_b1, gap_W2, gap_b2, te);
    k_cat2<<<cdiv(ROWS * 2 * DM, 256), 256, 0, stream>>>(h, te, xz);       // cat2 alias xz
    gemm(xz, 2 * DM, fuse_W, DM, fuse_b, h, DM, ROWS, DM, 2 * DM, 0);

    // ---- stage 3: 3 Mamba layers ----
    for (int l = 0; l < NL; ++l) {
        const float* ln_g      = (const float*)d_in[LBASE + l * LSTRIDE + 0];
        const float* ln_b      = (const float*)d_in[LBASE + l * LSTRIDE + 1];
        const float* in_proj   = (const float*)d_in[LBASE + l * LSTRIDE + 2];
        const float* conv_w    = (const float*)d_in[LBASE + l * LSTRIDE + 3];
        const float* conv_b    = (const float*)d_in[LBASE + l * LSTRIDE + 4];
        const float* x_proj    = (const float*)d_in[LBASE + l * LSTRIDE + 5];
        const float* dt_w      = (const float*)d_in[LBASE + l * LSTRIDE + 6];
        const float* dt_b      = (const float*)d_in[LBASE + l * LSTRIDE + 7];
        const float* A_log     = (const float*)d_in[LBASE + l * LSTRIDE + 8];
        const float* Dp        = (const float*)d_in[LBASE + l * LSTRIDE + 9];
        const float* out_proj  = (const float*)d_in[LBASE + l * LSTRIDE + 10];
        const float* log_decay = (const float*)d_in[LBASE + l * LSTRIDE + 11];

        k_ln<<<ROWS, 256, 0, stream>>>(h, hn, ln_g, ln_b, DM, 0);
        gemm(hn, DM, in_proj, 2 * DI, nullptr, xz, 2 * DI, ROWS, 2 * DI, DM, 0);
        k_conv<<<cdiv(ROWS * DI, 256), 256, 0, stream>>>(xz, conv_w, conv_b, xc);
        gemm(xc, DI, x_proj, DTR + 2 * DSTATE, nullptr, dbc, DTR + 2 * DSTATE,
             ROWS, DTR + 2 * DSTATE, DI, 0);
        gemm(dbc, DTR + 2 * DSTATE, dt_w, DI, dt_b, dtb, DI, ROWS, DI, DTR, 1); // softplus
        {
            dim3 g(DI / 256, BB);
            k_scan<<<g, 256, 0, stream>>>(dbc, dtb, xc, xz, A_log, Dp, ybuf);
        }
        gemm(ybuf, DI, out_proj, DM, nullptr, hm, DM, ROWS, DM, DI, 0);
        k_resid<<<cdiv(ROWS * DM, 256), 256, 0, stream>>>(h, hm, log_decay);
    }

    // ---- stage 4: final norm + pooling + projection ----
    k_ln<<<ROWS, 256, 0, stream>>>(h, hn, norm_g, norm_b, DM, 1);          // hf = hn
    gemm(hn, DM, pool_W, DM / 2, pool_Wb, xz, DM / 2, ROWS, DM / 2, DM, 2); // tanh tmp alias xz
    k_scores<<<ROWS, 128, 0, stream>>>(xz, pool_u, dbc);                    // scores alias dbc
    k_pool<<<BB, 256, 0, stream>>>(hn, dbc, xc);                            // cat4 alias xc
    gemm(xc, 4 * DM, pool_proj, DM, pool_proj_b, out, DM, BB, DM, 4 * DM, 0);
}
